// SK_71897752535112
// MI455X (gfx1250) — matrix operand order repaired and checked by reading, not compiled, not run on hardware
//
#include <hip/hip_runtime.h>
#include <hip/hip_bf16.h>

typedef __bf16 v16bf __attribute__((ext_vector_type(16)));
typedef float  v8f   __attribute__((ext_vector_type(8)));
typedef float  v4f   __attribute__((ext_vector_type(4)));
typedef int    vi4   __attribute__((vector_size(16)));   // matches builtin param type

#define C_DIM  1024
#define T_DIM  20480
#define NC_DIM 1024
#define K_SEL  3072

#define BMX 128     // output-channel rows per block
#define BNX 64      // time columns per block
#define KC  32      // K chunk (== WMMA K)
#define KP  40      // padded K stride in halves: 80B rows -> 16B aligned, conflict-free

union Frag { v16bf v; unsigned int u[8]; };

#if __has_builtin(__builtin_amdgcn_global_load_async_to_lds_b128)
#define HAVE_ASYNC_LDS 1
#else
#define HAVE_ASYNC_LDS 0
#endif

__device__ __forceinline__ void cp16_to_lds(const void* g, void* l)
{
#if HAVE_ASYNC_LDS
    __builtin_amdgcn_global_load_async_to_lds_b128((vi4*)g, (vi4*)l, 0, 0);
#else
    *(uint4*)l = *(const uint4*)g;
#endif
}

__device__ __forceinline__ void wait_async_all()
{
#if HAVE_ASYNC_LDS
#if __has_builtin(__builtin_amdgcn_s_wait_asynccnt)
    __builtin_amdgcn_s_wait_asynccnt(0);
#else
    asm volatile("s_wait_asynccnt 0" ::: "memory");
#endif
#endif
}

// ---------------------------------------------------------------------------
// Pre-pass 1: W (f32, [NC][C]) -> Whi/Wlo bf16 hi/lo, same layout
// ---------------------------------------------------------------------------
__global__ __launch_bounds__(256) void sk_wconv(
    const float* __restrict__ W, __bf16* __restrict__ Whi, __bf16* __restrict__ Wlo)
{
    const size_t idx = (size_t)(blockIdx.x * 256 + threadIdx.x) * 4;
    const v4f w = *(const v4f*)(W + idx);
    #pragma unroll
    for (int j = 0; j < 4; ++j) {
        float f = w[j];
        __bf16 h = (__bf16)f;
        Whi[idx + j] = h;
        Wlo[idx + j] = (__bf16)(f - (float)h);
    }
}

// ---------------------------------------------------------------------------
// Pre-pass 2: X (f32, [C][T]) -> XhiT/XloT bf16 hi/lo, TRANSPOSED [T][C]
// so the GEMM B operand has K contiguous (pure-copy LDS staging).
// ---------------------------------------------------------------------------
__global__ __launch_bounds__(256) void sk_xpose(
    const float* __restrict__ X, __bf16* __restrict__ XhiT, __bf16* __restrict__ XloT)
{
    __shared__ float tile[32][33];
    const int t0 = blockIdx.x * 32;
    const int c0 = blockIdx.y * 32;
    const int tl = threadIdx.x & 31;
    const int cl = threadIdx.x >> 5;   // 0..7
    #pragma unroll
    for (int j = 0; j < 4; ++j)
        tile[cl + 8 * j][tl] = X[(size_t)(c0 + cl + 8 * j) * T_DIM + t0 + tl];
    __syncthreads();
    #pragma unroll
    for (int j = 0; j < 4; ++j) {
        const int tr = cl + 8 * j;
        float f = tile[tl][tr];                      // X[c0+tl][t0+tr]
        __bf16 h = (__bf16)f;
        XhiT[(size_t)(t0 + tr) * C_DIM + c0 + tl] = h;
        XloT[(size_t)(t0 + tr) * C_DIM + c0 + tl] = (__bf16)(f - (float)h);
    }
}

// ---------------------------------------------------------------------------
// GEMM: H = relu(W @ X + b), fused partial score = w_score . H  (atomicAdd)
// Split-bf16 3-term WMMA; double-buffered LDS fed by async global->LDS copies.
// ---------------------------------------------------------------------------
__global__ __launch_bounds__(256) void sk_gemm(
    const __bf16* __restrict__ Whi, const __bf16* __restrict__ Wlo,
    const __bf16* __restrict__ XhiT, const __bf16* __restrict__ XloT,
    const float* __restrict__ bias, const float* __restrict__ wscore,
    float* __restrict__ H, float* __restrict__ scoreAcc)
{
    __shared__ __bf16 AsH[2][BMX * KP];   // 20480 B
    __shared__ __bf16 AsL[2][BMX * KP];   // 20480 B
    __shared__ __bf16 BsH[2][BNX * KP];   // 10240 B
    __shared__ __bf16 BsL[2][BNX * KP];   // 10240 B   (total 60 KB)

    const int tid    = threadIdx.x;
    const int lane   = tid & 31;
    const int wave   = tid >> 5;
    const int lane16 = lane & 15;
    const int hi     = lane >> 4;

    const int m0 = blockIdx.y * BMX;
    const int n0 = blockIdx.x * BNX;

    const int wm = wave >> 1;         // 0..3 : 32-row strip
    const int wn = wave & 1;          // 0..1 : 32-col strip

    v8f acc[2][2] = {};

    auto stage = [&](int kb, int p) {
        // A tiles: 128 rows x 32 halves, 2 b128 per thread per array
        #pragma unroll
        for (int i = 0; i < 2; ++i) {
            const int idx = tid + i * 256;
            const int row = idx >> 2;
            const int kg  = (idx & 3) * 8;
            const size_t so = (size_t)(m0 + row) * C_DIM + kb + kg;
            cp16_to_lds(Whi + so, &AsH[p][row * KP + kg]);
            cp16_to_lds(Wlo + so, &AsL[p][row * KP + kg]);
        }
        // B tiles: 64 rows x 32 halves, 1 b128 per thread per array
        {
            const int row = tid >> 2;
            const int kg  = (tid & 3) * 8;
            const size_t so = (size_t)(n0 + row) * C_DIM + kb + kg;
            cp16_to_lds(XhiT + so, &BsH[p][row * KP + kg]);
            cp16_to_lds(XloT + so, &BsL[p][row * KP + kg]);
        }
    };

    stage(0, 0);
    wait_async_all();
    __syncthreads();

    int p = 0;
    for (int kb = 0; kb < C_DIM; kb += KC) {
        const int nkb = kb + KC;
        if (nkb < C_DIM) stage(nkb, p ^ 1);   // async prefetch next chunk

        // ---- B fragments (32x16 bf16 layout: vgpr g holds K = (g<4 ? 2g : 16+2(g-4)) + 8*hi,+1) ----
        Frag bfr[2][2];
        #pragma unroll
        for (int nt = 0; nt < 2; ++nt) {
            const int n = wn * 32 + nt * 16 + lane16;
            #pragma unroll
            for (int g = 0; g < 8; ++g) {
                const int k = (g < 4 ? 2 * g : 16 + 2 * (g - 4)) + 8 * hi;
                bfr[nt][0].u[g] = *(const unsigned int*)&BsH[p][n * KP + k];
                bfr[nt][1].u[g] = *(const unsigned int*)&BsL[p][n * KP + k];
            }
        }
        // ---- per M-tile: A fragments + 3-term WMMA ----
        #pragma unroll
        for (int mt = 0; mt < 2; ++mt) {
            const int m = wm * 32 + mt * 16 + lane16;
            Frag ah, al;
            #pragma unroll
            for (int g = 0; g < 8; ++g) {
                const int k = (g < 4 ? 2 * g : 16 + 2 * (g - 4)) + 8 * hi;
                ah.u[g] = *(const unsigned int*)&AsH[p][m * KP + k];
                al.u[g] = *(const unsigned int*)&AsL[p][m * KP + k];
            }
            #pragma unroll
            for (int nt = 0; nt < 2; ++nt) {
                acc[mt][nt] = __builtin_amdgcn_wmma_f32_16x16x32_bf16(
                    false, ah.v, false, bfr[nt][0].v, (short)0, acc[mt][nt], false, false);
                acc[mt][nt] = __builtin_amdgcn_wmma_f32_16x16x32_bf16(
                    false, al.v, false, bfr[nt][0].v, (short)0, acc[mt][nt], false, false);
                acc[mt][nt] = __builtin_amdgcn_wmma_f32_16x16x32_bf16(
                    false, ah.v, false, bfr[nt][1].v, (short)0, acc[mt][nt], false, false);
            }
        }

        if (nkb < C_DIM) wait_async_all();
        __syncthreads();
        p ^= 1;
    }

    // ---- epilogue: bias + relu, store H, fused partial score reduction ----
    #pragma unroll
    for (int nt = 0; nt < 2; ++nt) {
        const int ng = n0 + wn * 32 + nt * 16 + lane16;
        float ps = 0.f;
        #pragma unroll
        for (int mt = 0; mt < 2; ++mt) {
            #pragma unroll
            for (int v = 0; v < 8; ++v) {
                const int mg = m0 + wm * 32 + mt * 16 + 8 * hi + v;   // C/D layout
                float val = acc[mt][nt][v] + bias[mg];
                val = fmaxf(val, 0.f);
                H[(size_t)mg * T_DIM + ng] = val;
                ps = fmaf(wscore[mg], val, ps);
            }
        }
        ps += __shfl_xor(ps, 16, 32);
        if (hi == 0) atomicAdd(&scoreAcc[ng], ps);
    }
}

// ---------------------------------------------------------------------------
__global__ __launch_bounds__(256) void sk_init(float* scoreAcc, unsigned int* cnt)
{
    int t = blockIdx.x * 256 + threadIdx.x;
    if (t < T_DIM) { scoreAcc[t] = 0.f; cnt[t] = 0u; }
}

__global__ __launch_bounds__(256) void sk_sigmoid(
    const float* __restrict__ scoreAcc, const float* __restrict__ bscore,
    float* __restrict__ scores)
{
    int t = blockIdx.x * 256 + threadIdx.x;
    if (t < T_DIM)
        scores[t] = 1.f / (1.f + __expf(-(scoreAcc[t] + bscore[0])));
}

// rank[t] = #{u: s[u] > s[t]} + #{u < t: s[u] == s[t]}   (lax.top_k tie order)
#define UCHUNK 2560
__global__ __launch_bounds__(256) void sk_count(
    const float* __restrict__ scores, unsigned int* __restrict__ cnt)
{
    __shared__ float su[UCHUNK];
    const int tid   = threadIdx.x;
    const int t     = blockIdx.x * 256 + tid;
    const int ubase = blockIdx.y * UCHUNK;
    for (int i = tid; i < UCHUNK; i += 256) su[i] = scores[ubase + i];
    __syncthreads();
    const float st = scores[t];
    unsigned int c = 0;
    for (int i = 0; i < UCHUNK; ++i) {
        float v = su[i];
        int   u = ubase + i;
        c += (unsigned int)((v > st) || ((v == st) && (u < t)));
    }
    atomicAdd(&cnt[t], c);
}

// single-block chunked prefix scan: picks[] = sorted indices with rank < k
__global__ __launch_bounds__(256) void sk_scan(
    const unsigned int* __restrict__ cnt, int* __restrict__ picks)
{
    __shared__ int sdata[256];
    __shared__ int sbase;
    const int tid = threadIdx.x;
    if (tid == 0) sbase = 0;
    __syncthreads();
    for (int b = 0; b < T_DIM; b += 256) {
        const int t    = b + tid;
        const int keep = (cnt[t] < (unsigned int)K_SEL) ? 1 : 0;
        sdata[tid] = keep;
        __syncthreads();
        for (int off = 1; off < 256; off <<= 1) {
            int v = (tid >= off) ? sdata[tid - off] : 0;
            __syncthreads();
            sdata[tid] += v;
            __syncthreads();
        }
        const int incl = sdata[tid];
        if (keep) picks[sbase + incl - 1] = t;
        __syncthreads();
        if (tid == 255) sbase += incl;
        __syncthreads();
    }
}

__global__ __launch_bounds__(256) void sk_gather(
    const float* __restrict__ H, const int* __restrict__ picks,
    float* __restrict__ out)
{
    const int j = blockIdx.y * 256 + threadIdx.x;  // 0..3071
    const int c = blockIdx.x;                      // 0..1023
    out[(size_t)c * K_SEL + j] = H[(size_t)c * T_DIM + picks[j]];
}

// ---------------------------------------------------------------------------
extern "C" void kernel_launch(void* const* d_in, const int* in_sizes, int n_in,
                              void* d_out, int out_size, void* d_ws, size_t ws_size,
                              hipStream_t stream)
{
    const float* x       = (const float*)d_in[0];
    const float* w_fcsn  = (const float*)d_in[1];
    const float* b_fcsn  = (const float*)d_in[2];
    const float* w_score = (const float*)d_in[3];
    const float* b_score = (const float*)d_in[4];
    float* out = (float*)d_out;

    float*        H        = (float*)d_ws;                       // 1024*20480 f32
    float*        scoreAcc = H + (size_t)NC_DIM * T_DIM;         // 20480 f32
    float*        scores   = scoreAcc + T_DIM;                   // 20480 f32
    unsigned int* cnt      = (unsigned int*)(scores + T_DIM);    // 20480 u32
    int*          picks    = (int*)(cnt + T_DIM);                // 3072 i32
    __bf16*       Whi      = (__bf16*)(picks + K_SEL);           // 1M bf16
    __bf16*       Wlo      = Whi + (size_t)NC_DIM * C_DIM;       // 1M bf16
    __bf16*       XhiT     = Wlo + (size_t)NC_DIM * C_DIM;       // 20480*1024 bf16
    __bf16*       XloT     = XhiT + (size_t)T_DIM * C_DIM;       // 20480*1024 bf16

    sk_init   <<<dim3(80),                       256, 0, stream>>>(scoreAcc, cnt);
    sk_wconv  <<<dim3(NC_DIM * C_DIM / 1024),    256, 0, stream>>>(w_fcsn, Whi, Wlo);
    sk_xpose  <<<dim3(T_DIM / 32, C_DIM / 32),   256, 0, stream>>>(x, XhiT, XloT);
    sk_gemm   <<<dim3(T_DIM / BNX, NC_DIM / BMX), 256, 0, stream>>>(Whi, Wlo, XhiT, XloT,
                                                                    b_fcsn, w_score, H, scoreAcc);
    sk_sigmoid<<<dim3(80),                       256, 0, stream>>>(scoreAcc, b_score, scores);
    sk_count  <<<dim3(80, 8),                    256, 0, stream>>>(scores, cnt);
    sk_scan   <<<dim3(1),                        256, 0, stream>>>(cnt, picks);
    sk_gather <<<dim3(NC_DIM, K_SEL / 256),      256, 0, stream>>>(H, picks, out);
}
